// ProtoNet_11974368821656
// MI455X (gfx1250) — matrix operand order repaired and checked by reading, not compiled, not run on hardware
//
#include <hip/hip_runtime.h>
#include <hip/hip_bf16.h>
#include <math.h>

// ---------------- problem constants (from reference) ----------------
#define N_TOT 6144          // nk*P + nq*P = 4096 + 2048
#define C_DIM 320           // channels
#define NS    4096          // support columns
#define NQ    2048          // query columns
#define SIGMA 0.25f
#define ALPHA 0.99f
#define TOPK  20
#define EPS_F 2.2204460492503131e-16f   // np.finfo(float64).eps, as f32
#define CHEB_ITERS 120

typedef __attribute__((ext_vector_type(16))) __bf16 v16bf;
typedef __attribute__((ext_vector_type(8)))  float  v8f;

union Frag { unsigned u32[8]; v16bf v; };

// ---------------- kernel 1: pack X to bf16 + row squared norms ----------------
// emb_all[c][n] = n<NS ? sup[c*NS+n] : qry[c*NQ+(n-NS)]   (pure flat reshape)
// X[n][c] = emb_all[c][n] / (SIGMA+eps);  Xb row-major [N][C] bf16 (RNE), sq[n]=|X_n|^2
__global__ void pack_kernel(const float* __restrict__ sup, const float* __restrict__ qry,
                            unsigned short* __restrict__ Xb, float* __restrict__ sq) {
  const int n = blockIdx.x;
  const int tid = threadIdx.x;                 // 64 threads
  __shared__ float red[64];
  const float inv = 1.0f / (SIGMA + EPS_F);
  float partial = 0.0f;
  for (int c = tid; c < C_DIM; c += 64) {
    float v = (n < NS) ? sup[(size_t)c * NS + n] : qry[(size_t)c * NQ + (n - NS)];
    v *= inv;
    partial += v * v;
    unsigned u = __float_as_uint(v);
    u += 0x7FFFu + ((u >> 16) & 1u);           // round-to-nearest-even bf16
    Xb[(size_t)n * C_DIM + c] = (unsigned short)(u >> 16);
  }
  red[tid] = partial;
  __syncthreads();
  for (int s = 32; s > 0; s >>= 1) { if (tid < s) red[tid] += red[tid + s]; __syncthreads(); }
  if (tid == 0) sq[n] = red[0];
}

// ---------------- kernel 2: WMMA bf16 GEMM -> affinity W = exp(-d2/2) ----------------
// One wave computes one 16x16 tile of G = X X^T; block = 8 waves = 16 rows x 128 cols.
__global__ void __launch_bounds__(256)
wmma_affinity_kernel(const unsigned short* __restrict__ Xb,
                     const float* __restrict__ sq,
                     float* __restrict__ W) {
  const int lane = threadIdx.x & 31;
  const int wave = threadIdx.x >> 5;                       // 0..7
  const int rowBase = blockIdx.y * 16;
  const int colBase = (blockIdx.x * 8 + wave) * 16;
  const int half = lane >> 4;                              // lane group
  const int m    = lane & 15;

  v8f acc = {};
#pragma unroll
  for (int kb = 0; kb < C_DIM; kb += 32) {
    Frag a, b;
#pragma unroll
    for (int v = 0; v < 8; ++v) {
      // A 16x32 bf16 layout: lanes0-15: V0..3 K=0..7, V4..7 K=16..23 (pairs);
      //                      lanes16-31: V0..3 K=8..15, V4..7 K=24..31
      const int ka = (v >> 2) * 16 + half * 8 + (v & 3) * 2;
      a.u32[v] = *(const unsigned*)(Xb + (size_t)(rowBase + m) * C_DIM + kb + ka);
      // B 32x16 bf16 (same K layout as A): lanes0-15 K=0..7,16..23; lanes16-31 K=8..15,24..31
      const int kbi = (v >> 2) * 16 + half * 8 + (v & 3) * 2;
      b.u32[v] = *(const unsigned*)(Xb + (size_t)(colBase + m) * C_DIM + kb + kbi);
    }
    acc = __builtin_amdgcn_wmma_f32_16x16x32_bf16(false, a.v, false, b.v,
                                                  (short)0, acc, false, false);
  }

  const float invC = 1.0f / (float)C_DIM;
  const float sqj  = sq[colBase + m];                      // column j = colBase + (lane&15)
#pragma unroll
  for (int v = 0; v < 8; ++v) {
    const int i = rowBase + v + half * 8;                  // C/D layout: VGPRv -> M=v (+8 hi half)
    const int j = colBase + m;
    const float d2 = (sq[i] + sqj - 2.0f * acc[v]) * invC;
    W[(size_t)i * N_TOT + j] = __expf(-0.5f * d2);
  }
}

// ---------------- kernel 3: per-row top-k threshold (peel 20 maxima) ----------------
__global__ void topk_thresh_kernel(const float* __restrict__ W, float* __restrict__ t) {
  const int i = blockIdx.x;
  const int tid = threadIdx.x;                 // 256
  __shared__ float red[256];
  const float* row = W + (size_t)i * N_TOT;
  float cur = 3.4e38f;
  for (int rep = 0; rep < TOPK; ++rep) {
    float local = -3.4e38f;
    for (int j = tid; j < N_TOT; j += 256) {
      const float w = row[j];
      if (w < cur && w > local) local = w;
    }
    red[tid] = local;
    __syncthreads();
    for (int s = 128; s > 0; s >>= 1) { if (tid < s) red[tid] = fmaxf(red[tid], red[tid + s]); __syncthreads(); }
    cur = red[0];
    __syncthreads();
  }
  if (tid == 0) t[i] = cur;
}

// ---------------- kernel 4: symmetric top-k mask + row sums (degrees) ----------------
__global__ void mask_rowsum_kernel(float* __restrict__ W, const float* __restrict__ t,
                                   float* __restrict__ D) {
  const int i = blockIdx.x;
  const int tid = threadIdx.x;                 // 256
  __shared__ float red[256];
  const float ti = t[i];
  float sum = 0.0f;
  for (int j = tid; j < N_TOT; j += 256) {
    float w        = W[(size_t)i * N_TOT + j];
    const float wt = W[(size_t)j * N_TOT + i];   // benign race: overwrite is keep-or-zero
    const bool keep = (w >= ti) || (wt >= t[j]);
    w = keep ? w : 0.0f;
    W[(size_t)i * N_TOT + j] = w;
    sum += w;
  }
  red[tid] = sum;
  __syncthreads();
  for (int s = 128; s > 0; s >>= 1) { if (tid < s) red[tid] += red[tid + s]; __syncthreads(); }
  if (tid == 0) D[i] = red[0];
}

// ---------------- kernel 5: D -> d^{-1/2} in place ----------------
__global__ void dinv_kernel(float* __restrict__ D) {
  const int n = blockIdx.x * blockDim.x + threadIdx.x;
  if (n < N_TOT) D[n] = rsqrtf(D[n] + EPS_F);
}

// ---------------- kernel 6: W <- alpha * dinv_i * W * dinv_j  (= alpha*S) ----------------
__global__ void scale_kernel(float* __restrict__ W, const float* __restrict__ dinv) {
  const size_t total = (size_t)N_TOT * N_TOT;
  for (size_t idx = (size_t)blockIdx.x * blockDim.x + threadIdx.x; idx < total;
       idx += (size_t)gridDim.x * blockDim.x) {
    const size_t i = idx / N_TOT, j = idx % N_TOT;
    W[idx] *= ALPHA * dinv[i] * dinv[j];
  }
}

// ---------------- kernel 7: init y, x1, d0 for Chebyshev (x0=0, theta=1) ----------------
__global__ void init_vec_kernel(const float* __restrict__ sup_y, float* __restrict__ y,
                                float* __restrict__ x, float* __restrict__ d) {
  const int n = blockIdx.x * blockDim.x + threadIdx.x;
  if (n < N_TOT) {
    const float v = (n < NS) ? sup_y[n] : 0.0f;
    y[n] = v;
    x[n] = v;     // x1 = x0 + (1/theta) r0 = y
    d[n] = v;     // d0 = (1/theta) r0
  }
}

// ---------------- kernel 8: one Chebyshev step ----------------
// A = I - alpha*S ; Wn holds alpha*S. r = y - x + Wn x ; d' = c1*d + c2*r ; x' = x + d'
__global__ void cheb_iter_kernel(const float* __restrict__ Wn, const float* __restrict__ y,
                                 const float* __restrict__ xin, const float* __restrict__ din,
                                 float* __restrict__ xout, float* __restrict__ dout,
                                 float c1, float c2) {
  const int i = blockIdx.x;
  const int tid = threadIdx.x;                 // 256
  __shared__ float red[256];
  const float4* row = (const float4*)(Wn + (size_t)i * N_TOT);
  const float4* x4  = (const float4*)xin;
  float sum = 0.0f;
  for (int j = tid; j < N_TOT / 4; j += 256) {
    const float4 a = row[j];
    const float4 b = x4[j];
    sum += a.x * b.x + a.y * b.y + a.z * b.z + a.w * b.w;
  }
  red[tid] = sum;
  __syncthreads();
  for (int s = 128; s > 0; s >>= 1) { if (tid < s) red[tid] += red[tid + s]; __syncthreads(); }
  if (tid == 0) {
    const float Sx = red[0];
    const float r  = y[i] - xin[i] + Sx;
    const float dn = c1 * din[i] + c2 * r;
    dout[i] = dn;
    xout[i] = xin[i] + dn;
  }
}

// ---------------- kernel 9: emit Fq = F[NS : NS+NQ] ----------------
__global__ void copy_out_kernel(const float* __restrict__ F, float* __restrict__ out) {
  const int n = blockIdx.x * blockDim.x + threadIdx.x;
  if (n < NQ) out[n] = F[NS + n];
}

// ---------------- host orchestration ----------------
extern "C" void kernel_launch(void* const* d_in, const int* in_sizes, int n_in,
                              void* d_out, int out_size, void* d_ws, size_t ws_size,
                              hipStream_t stream) {
  const float* sup   = (const float*)d_in[0];   // (4,320,1024) f32
  const float* qry   = (const float*)d_in[1];   // (2,320,1024) f32
  const float* sup_y = (const float*)d_in[2];   // (2,2,1024)   f32
  // d_in[3] (query_y) unused by the computation
  float* out = (float*)d_out;

  // workspace layout (floats)
  float* ws = (float*)d_ws;
  size_t off = 0;
  float* W            = ws + off; off += (size_t)N_TOT * N_TOT;        // 151 MB
  unsigned short* Xb  = (unsigned short*)(ws + off); off += (size_t)N_TOT * C_DIM / 2;
  float* sq  = ws + off; off += N_TOT;
  float* t   = ws + off; off += N_TOT;
  float* Dg  = ws + off; off += N_TOT;
  float* yv  = ws + off; off += N_TOT;
  float* x0  = ws + off; off += N_TOT;
  float* x1  = ws + off; off += N_TOT;
  float* db0 = ws + off; off += N_TOT;
  float* db1 = ws + off; off += N_TOT;

  // 1) pack + norms
  pack_kernel<<<N_TOT, 64, 0, stream>>>(sup, qry, Xb, sq);

  // 2) WMMA affinity matrix
  dim3 ggrid(N_TOT / 128, N_TOT / 16);
  wmma_affinity_kernel<<<ggrid, 256, 0, stream>>>(Xb, sq, W);

  // 3) top-k thresholds
  topk_thresh_kernel<<<N_TOT, 256, 0, stream>>>(W, t);

  // 4) mask + degrees
  mask_rowsum_kernel<<<N_TOT, 256, 0, stream>>>(W, t, Dg);

  // 5) degree^-1/2
  dinv_kernel<<<(N_TOT + 255) / 256, 256, 0, stream>>>(Dg);

  // 6) W <- alpha*S
  scale_kernel<<<18432, 256, 0, stream>>>(W, Dg);

  // 7) Chebyshev init (x0 = 0 handled analytically)
  init_vec_kernel<<<(N_TOT + 255) / 256, 256, 0, stream>>>(sup_y, yv, x0, db0);

  // 8) Chebyshev semi-iteration on spectrum [1-ALPHA, 1+ALPHA]
  const double theta = 1.0, delta = (double)ALPHA;
  const double sigma1 = theta / delta;
  double rho = delta / theta;                   // rho_0
  float* xs[2] = {x0, x1};
  float* ds[2] = {db0, db1};
  int cur = 0;
  for (int k = 1; k <= CHEB_ITERS; ++k) {
    const double rho_new = 1.0 / (2.0 * sigma1 - rho);
    const float c1 = (float)(rho_new * rho);
    const float c2 = (float)(2.0 * rho_new / delta);
    cheb_iter_kernel<<<N_TOT, 256, 0, stream>>>(W, yv, xs[cur], ds[cur],
                                                xs[cur ^ 1], ds[cur ^ 1], c1, c2);
    rho = rho_new;
    cur ^= 1;
  }

  // 9) output query rows
  copy_out_kernel<<<(NQ + 255) / 256, 256, 0, stream>>>(xs[cur], out);
}
